// FocusA_49881750176053
// MI455X (gfx1250) — matrix operand order repaired and checked by reading, not compiled, not run on hardware
//
#include <hip/hip_runtime.h>

// ---------------------------------------------------------------------------
// CDNA5 / gfx1250: WMMA bf16 GEMMs + flash attention.
// - v_wmma_f32_16x16x32_bf16 everywhere (wave32, fp32 accum)
// - GEMM: W tile staged to LDS via global_load_async_to_lds_b128 (ASYNCcnt),
//   double buffered; A fragments software-pipelined in registers.
// - Attention: V fragments loaded after softmax (overlap P LDS round-trip),
//   __launch_bounds__ sized to avoid VGPR spills.
// ---------------------------------------------------------------------------

typedef __attribute__((ext_vector_type(16))) __bf16 bf16x16;
typedef __attribute__((ext_vector_type(8)))  float  f32x8;

constexpr int Bb = 4, Ss = 1024, Dd = 1024, Hh = 16, HD = 64;
constexpr int M_ROWS = Bb * Ss;   // 4096 token rows

__device__ __forceinline__ __bf16 f2bf(float f) {
  unsigned u = __builtin_bit_cast(unsigned, f);
  unsigned r = (u + 0x7FFFu + ((u >> 16) & 1u)) >> 16;   // RNE
  unsigned short h = (unsigned short)r;
  return __builtin_bit_cast(__bf16, h);
}

// CDNA5 async global->LDS copy (16B per lane), tracked by ASYNCcnt.
__device__ __forceinline__ void async_ld16(unsigned lds_addr, const void* gptr) {
  asm volatile("global_load_async_to_lds_b128 %0, %1, off"
               :: "v"(lds_addr), "v"(gptr) : "memory");
}
__device__ __forceinline__ void wait_async0() {
  asm volatile("s_wait_asynccnt 0x0" ::: "memory");
}

// --- WMMA fragment loaders (contiguous k = 16*(lane/16) + e, same in A and B, wave32) ---------------------

// A (16x32 bf16): lane m = lane%16; k = (lane>=16?16:0) + e, e=0..15
__device__ __forceinline__ bf16x16 a_frag(const __bf16* mat, int row0, int rowmax,
                                          int k0, int ld) {
  const int lane = threadIdx.x & 31;
  int m = row0 + (lane & 15);
  m = m > rowmax ? rowmax : m;
  const int kb = k0 + ((lane >> 4) << 4);
  const __bf16* p = mat + (size_t)m * ld + kb;
  bf16x16 r;
#pragma unroll
  for (int e = 0; e < 8; ++e) r[e] = p[e];
#pragma unroll
  for (int e = 0; e < 8; ++e) r[8 + e] = p[8 + e];
  return r;
}

// B (32x16): stored matrix has N rows with K contiguous (W[N][K] or K[S][D]).
__device__ __forceinline__ bf16x16 b_frag_kcontig(const __bf16* mat, int n0, int nmax,
                                                  int k0, int ld) {
  const int lane = threadIdx.x & 31;
  int n = n0 + (lane & 15);
  n = n > nmax ? nmax : n;
  const int kk = k0 + ((lane >> 4) << 4);
  const __bf16* p = mat + (size_t)n * ld + kk;
  bf16x16 r;
#pragma unroll
  for (int e = 0; e < 16; ++e) r[e] = p[e];
  return r;
}

// B (32x16) where K runs across rows of the stored matrix (V[S][D]).
__device__ __forceinline__ bf16x16 b_frag_kstrided(const __bf16* mat, int krow0,
                                                   int krowmax, int n0, int ld) {
  const int lane = threadIdx.x & 31;
  const int n = n0 + (lane & 15);
  const int kk0 = krow0 + ((lane >> 4) << 4);
  bf16x16 r;
#pragma unroll
  for (int e = 0; e < 16; ++e) {
    int kr = kk0 + e; kr = kr > krowmax ? krowmax : kr;
    r[e] = mat[(size_t)kr * ld + n];
  }
  return r;
}

// --- LayerNorm (both param sets in one pass over x) -------------------------
__global__ void ln_kernel(const float* __restrict__ x,
                          const float* __restrict__ ga, const float* __restrict__ ba,
                          const float* __restrict__ gb, const float* __restrict__ bb,
                          __bf16* __restrict__ localbf, __bf16* __restrict__ globebf) {
  __shared__ float red[256];
  const int row = blockIdx.x;
  const float* xr = x + (size_t)row * Dd;
  float vals[4]; float s = 0.f;
#pragma unroll
  for (int i = 0; i < 4; ++i) { vals[i] = xr[threadIdx.x + i * 256]; s += vals[i]; }
  red[threadIdx.x] = s; __syncthreads();
  for (int off = 128; off > 0; off >>= 1) {
    if (threadIdx.x < off) red[threadIdx.x] += red[threadIdx.x + off];
    __syncthreads();
  }
  const float mean = red[0] * (1.f / Dd);
  __syncthreads();
  float s2 = 0.f;
#pragma unroll
  for (int i = 0; i < 4; ++i) { float d = vals[i] - mean; s2 += d * d; }
  red[threadIdx.x] = s2; __syncthreads();
  for (int off = 128; off > 0; off >>= 1) {
    if (threadIdx.x < off) red[threadIdx.x] += red[threadIdx.x + off];
    __syncthreads();
  }
  const float rstd = rsqrtf(red[0] * (1.f / Dd) + 1e-5f);
#pragma unroll
  for (int i = 0; i < 4; ++i) {
    const int c = threadIdx.x + i * 256;
    const float n = (vals[i] - mean) * rstd;
    localbf[(size_t)row * Dd + c] = f2bf(n * ga[c] + ba[c]);
    globebf[(size_t)row * Dd + c] = f2bf(n * gb[c] + bb[c]);
  }
}

__global__ void cvt_bf16_kernel(const float* __restrict__ src, __bf16* __restrict__ dst, int n) {
  const int i = blockIdx.x * 256 + threadIdx.x;
  if (i < n) dst[i] = f2bf(src[i]);
}

// --- WMMA GEMM: C[M,N] = A[M,K] @ W[N,K]^T + bias ---------------------------
// 8 waves; block tile 128x64. W tile (64x32) shared by all waves -> staged in
// LDS via async copy (double buffered); A fragments pipelined in registers.
template <int KC>
__global__ void gemm_bf16_kernel(const __bf16* __restrict__ A, int lda,
                                 const __bf16* __restrict__ W, int ldw,
                                 const float* __restrict__ bias,
                                 float* __restrict__ Cf, int ldcf,
                                 __bf16* __restrict__ Cb, int ldcb) {
  __shared__ __bf16 wtile[2][64 * 32];          // 2 x 4KB
  const int wave = threadIdx.x >> 5;
  const int lane = threadIdx.x & 31;
  const int row0 = blockIdx.y * 128 + wave * 16;
  const int col0 = blockIdx.x * 64;

  auto stage = [&](int buf, int k0) {           // 256 thr x 16B = 4KB tile
    const int r = threadIdx.x >> 2;             // 0..63  (N row)
    const int kc = (threadIdx.x & 3) * 8;       // 0/8/16/24
    const __bf16* g = W + (size_t)(col0 + r) * ldw + k0 + kc;
    async_ld16((unsigned)(uintptr_t)&wtile[buf][r * 32 + kc], g);
  };
  auto b_lds = [&](int buf, int t) {            // B frag from staged tile
    const int nl = t * 16 + (lane & 15);
    const int kk = (lane >> 4) << 4;
    const __bf16* p = &wtile[buf][nl * 32 + kk];
    bf16x16 r;
#pragma unroll
    for (int e = 0; e < 16; ++e) r[e] = p[e];
    return r;
  };

  f32x8 acc[4];
#pragma unroll
  for (int t = 0; t < 4; ++t)
#pragma unroll
    for (int v = 0; v < 8; ++v) acc[t][v] = 0.f;

  stage(0, 0);
  wait_async0();
  __syncthreads();
  bf16x16 a_cur = a_frag(A, row0, M_ROWS - 1, 0, lda);

  for (int k0 = 0; k0 < KC; k0 += 32) {
    const int buf = (k0 >> 5) & 1;
    const bool more = (k0 + 32) < KC;
    if (more) stage(buf ^ 1, k0 + 32);          // async prefetch next W tile
    bf16x16 b0 = b_lds(buf, 0), b1 = b_lds(buf, 1);
    bf16x16 b2 = b_lds(buf, 2), b3 = b_lds(buf, 3);
    bf16x16 a_next = more ? a_frag(A, row0, M_ROWS - 1, k0 + 32, lda) : a_cur;
    acc[0] = __builtin_amdgcn_wmma_f32_16x16x32_bf16(false, a_cur, false, b0, (short)0, acc[0], false, false);
    acc[1] = __builtin_amdgcn_wmma_f32_16x16x32_bf16(false, a_cur, false, b1, (short)0, acc[1], false, false);
    acc[2] = __builtin_amdgcn_wmma_f32_16x16x32_bf16(false, a_cur, false, b2, (short)0, acc[2], false, false);
    acc[3] = __builtin_amdgcn_wmma_f32_16x16x32_bf16(false, a_cur, false, b3, (short)0, acc[3], false, false);
    a_cur = a_next;
    if (more) { wait_async0(); __syncthreads(); }
  }

  const int half = lane >> 4, n = lane & 15;
#pragma unroll
  for (int t = 0; t < 4; ++t) {
    const int col = col0 + t * 16 + n;
    const float bi = bias ? bias[col] : 0.f;
#pragma unroll
    for (int v = 0; v < 8; ++v) {
      const int m = row0 + v + half * 8;
      const float val = acc[t][v] + bi;
      if (Cf) Cf[(size_t)m * ldcf + col] = val;
      if (Cb) Cb[(size_t)m * ldcb + col] = f2bf(val);
    }
  }
}

// --- Causal global attention (flash-style, per head) ------------------------
__global__ __launch_bounds__(256, 1)
void attn_global_kernel(const __bf16* __restrict__ qb_,
                        const __bf16* __restrict__ kb_,
                        const __bf16* __restrict__ vb_,
                        __bf16* __restrict__ ob_) {
  __shared__ __bf16 ldsP[8 * 512];
  const int wave = threadIdx.x >> 5, lane = threadIdx.x & 31;
  const int half = lane >> 4, nn = lane & 15;
  const int bh = blockIdx.y, b = bh / Hh, h = bh % Hh, c0 = h * HD;
  const __bf16* qm = qb_ + (size_t)b * Ss * Dd;
  const __bf16* km = kb_ + (size_t)b * Ss * Dd;
  const __bf16* vm = vb_ + (size_t)b * Ss * Dd;
  const int q0 = blockIdx.x * 128 + wave * 16;
  bf16x16 aq0 = a_frag(qm, q0, Ss - 1, c0, Dd);
  bf16x16 aq1 = a_frag(qm, q0, Ss - 1, c0 + 32, Dd);
  f32x8 o[4]; float mr[8], lr[8];
#pragma unroll
  for (int t = 0; t < 4; ++t)
#pragma unroll
    for (int v = 0; v < 8; ++v) o[t][v] = 0.f;
#pragma unroll
  for (int v = 0; v < 8; ++v) { mr[v] = -1e30f; lr[v] = 0.f; }
  const float scale = 0.125f;                       // 1/sqrt(HD)
  const int nkb = (q0 + 15) / 32 + 1;
  __bf16* myP = ldsP + wave * 512;
  for (int ik = 0; ik < nkb; ++ik) {
    const int kbase = ik * 32;
    f32x8 s0, s1;
#pragma unroll
    for (int v = 0; v < 8; ++v) { s0[v] = 0.f; s1[v] = 0.f; }
    s0 = __builtin_amdgcn_wmma_f32_16x16x32_bf16(false, aq0, false,
          b_frag_kcontig(km, kbase, Ss - 1, c0, Dd), (short)0, s0, false, false);
    s0 = __builtin_amdgcn_wmma_f32_16x16x32_bf16(false, aq1, false,
          b_frag_kcontig(km, kbase, Ss - 1, c0 + 32, Dd), (short)0, s0, false, false);
    s1 = __builtin_amdgcn_wmma_f32_16x16x32_bf16(false, aq0, false,
          b_frag_kcontig(km, kbase + 16, Ss - 1, c0, Dd), (short)0, s1, false, false);
    s1 = __builtin_amdgcn_wmma_f32_16x16x32_bf16(false, aq1, false,
          b_frag_kcontig(km, kbase + 16, Ss - 1, c0 + 32, Dd), (short)0, s1, false, false);
    float p0[8], p1[8];
#pragma unroll
    for (int v = 0; v < 8; ++v) {
      const int qrow = q0 + v + half * 8;
      float x0 = s0[v] * scale, x1 = s1[v] * scale;
      if (kbase + nn > qrow)      x0 = -1e30f;      // causal mask
      if (kbase + 16 + nn > qrow) x1 = -1e30f;
      float tm = fmaxf(x0, x1);
      tm = fmaxf(tm, __shfl_xor(tm, 1)); tm = fmaxf(tm, __shfl_xor(tm, 2));
      tm = fmaxf(tm, __shfl_xor(tm, 4)); tm = fmaxf(tm, __shfl_xor(tm, 8));
      const float nm = fmaxf(mr[v], tm);
      const float al = __expf(mr[v] - nm);
      const float e0 = __expf(x0 - nm), e1 = __expf(x1 - nm);
      float rs = e0 + e1;
      rs += __shfl_xor(rs, 1); rs += __shfl_xor(rs, 2);
      rs += __shfl_xor(rs, 4); rs += __shfl_xor(rs, 8);
      lr[v] = lr[v] * al + rs; mr[v] = nm;
      p0[v] = e0; p1[v] = e1;
#pragma unroll
      for (int t = 0; t < 4; ++t) o[t][v] *= al;
    }
#pragma unroll
    for (int v = 0; v < 8; ++v) {                   // P -> LDS (16x32 row-major)
      const int r = v + half * 8;
      myP[r * 32 + nn]      = f2bf(p0[v]);
      myP[r * 32 + 16 + nn] = f2bf(p1[v]);
    }
    // V fragments issued here: global loads overlap the P LDS round-trip,
    // and s0/s1/p0/p1 are dead so liveness stays within budget.
    bf16x16 bv[4];
#pragma unroll
    for (int t = 0; t < 4; ++t) bv[t] = b_frag_kstrided(vm, kbase, Ss - 1, c0 + t * 16, Dd);
    bf16x16 pa;
    {
      const int m = lane & 15, kb2 = (lane >> 4) << 4;
#pragma unroll
      for (int e = 0; e < 8; ++e) pa[e] = myP[m * 32 + kb2 + e];
#pragma unroll
      for (int e = 0; e < 8; ++e) pa[8 + e] = myP[m * 32 + kb2 + 8 + e];
    }
#pragma unroll
    for (int t = 0; t < 4; ++t)
      o[t] = __builtin_amdgcn_wmma_f32_16x16x32_bf16(false, pa, false, bv[t],
                                                     (short)0, o[t], false, false);
  }
#pragma unroll
  for (int v = 0; v < 8; ++v) {
    const float inv = 1.f / lr[v];
    const int qrow = q0 + v + half * 8;
#pragma unroll
    for (int t = 0; t < 4; ++t)
      ob_[((size_t)b * Ss + qrow) * Dd + c0 + t * 16 + nn] = f2bf(o[t][v] * inv);
  }
}

// --- Span predictor ---------------------------------------------------------
__global__ void colmean_kernel(const float* __restrict__ go, float* __restrict__ cm) {
  const int idx = blockIdx.x * 256 + threadIdx.x;
  const int b = idx >> 10, d = idx & 1023;
  float s = 0.f;
  const float* p = go + (size_t)b * Ss * Dd + d;
  for (int i = 0; i < Ss; ++i) s += p[(size_t)i * Dd];
  cm[idx] = s * (1.f / Ss);
}

__global__ void predict_kernel(const float* __restrict__ cm, const float* __restrict__ Wp,
                               const float* __restrict__ bp,
                               float* __restrict__ pf, int* __restrict__ pi) {
  __shared__ float red[256];
  float sm_acc = 0.f;
  for (int b = 0; b < Bb; ++b) {
    float s = 0.f;
    for (int d = threadIdx.x; d < Dd; d += 256) s += cm[b * Dd + d] * Wp[d];
    red[threadIdx.x] = s; __syncthreads();
    for (int off = 128; off > 0; off >>= 1) {
      if (threadIdx.x < off) red[threadIdx.x] += red[threadIdx.x + off];
      __syncthreads();
    }
    const float dot = red[0] + bp[0];
    __syncthreads();
    float sc = 1.f / (1.f + __expf(-dot));
    sc = fminf(fmaxf(sc, 0.f), 1.f);
    sm_acc += sc;
  }
  const float smean = sm_acc * (1.f / Bb);
  if (threadIdx.x == 0) {
    pf[0] = smean;
    int win = (int)(256.f * smean); if (win < 1) win = 1;
    int sl  = (int)(512.f * smean); if (sl  < 1) sl  = 1;
    int lm  = 512; if (sl < lm) lm = sl; if (win < lm) lm = win;
    pi[0] = win; pi[1] = sl; pi[2] = lm;
  }
}

// --- Adaptive-span windowed local attention (data-dependent, early-exit) ----
__global__ __launch_bounds__(256, 1)
void attn_local_kernel(const __bf16* __restrict__ lb_, __bf16* __restrict__ comb,
                       const float* __restrict__ pf, const int* __restrict__ pi) {
  __shared__ __bf16 ldsP[8 * 512];
  const int win = pi[0], span_len = pi[1], local_max = pi[2];
  const float sm = pf[0];
  const int st = blockIdx.x * win;
  if (st >= Ss) return;
  const int en = st + win < Ss ? st + win : Ss;
  const int wlen = en - st;
  int ks = st - span_len + win; if (ks < 0) ks = 0;
  int ke = st + span_len; if (ke > Ss) ke = Ss;
  const int klen = ke - ks;
  int eff = (int)((float)wlen * sm);
  if (eff > wlen) eff = wlen;
  if (eff > klen) eff = klen;
  if (eff > local_max) eff = local_max;
  const int bh = blockIdx.y, b = bh / Hh, h = bh % Hh, c0 = h * HD;
  const __bf16* lm = lb_ + (size_t)b * Ss * Dd;
  __bf16* cmb = comb + (size_t)b * Ss * (2 * Dd);
  const int effz = eff > 0 ? eff : 0;
  const __bf16 z = f2bf(0.f);
  for (int idx = threadIdx.x; idx < (wlen - effz) * HD; idx += 256) {
    const int r = st + effz + idx / HD, c = idx % HD;
    cmb[(size_t)r * (2 * Dd) + c0 + c] = z;                 // tail padding
  }
  if (eff <= 0) return;
  const float temp = 1.f + 0.01f * (1.f - sm);
  const float scale = 0.35355339f / temp;                   // HD^-0.25 / temp
  const int wave = threadIdx.x >> 5, lane = threadIdx.x & 31;
  const int half = lane >> 4, nn = lane & 15;
  __bf16* myP = ldsP + wave * 512;
  const int ntq = (eff + 15) >> 4, nkb = (eff + 31) >> 5;
  const int kmaxrow = ks + eff - 1;
  for (int qt = wave; qt < ntq; qt += 8) {
    const int q0 = st + qt * 16;
    bf16x16 aq0 = a_frag(lm, q0, Ss - 1, c0, Dd);
    bf16x16 aq1 = a_frag(lm, q0, Ss - 1, c0 + 32, Dd);
    f32x8 o[4]; float mr[8], lr[8];
#pragma unroll
    for (int t = 0; t < 4; ++t)
#pragma unroll
      for (int v = 0; v < 8; ++v) o[t][v] = 0.f;
#pragma unroll
    for (int v = 0; v < 8; ++v) { mr[v] = -1e30f; lr[v] = 0.f; }
    for (int ik = 0; ik < nkb; ++ik) {
      const int kl = ik * 32;
      f32x8 s0, s1;
#pragma unroll
      for (int v = 0; v < 8; ++v) { s0[v] = 0.f; s1[v] = 0.f; }
      s0 = __builtin_amdgcn_wmma_f32_16x16x32_bf16(false, aq0, false,
            b_frag_kcontig(lm, ks + kl, kmaxrow, c0, Dd), (short)0, s0, false, false);
      s0 = __builtin_amdgcn_wmma_f32_16x16x32_bf16(false, aq1, false,
            b_frag_kcontig(lm, ks + kl, kmaxrow, c0 + 32, Dd), (short)0, s0, false, false);
      s1 = __builtin_amdgcn_wmma_f32_16x16x32_bf16(false, aq0, false,
            b_frag_kcontig(lm, ks + kl + 16, kmaxrow, c0, Dd), (short)0, s1, false, false);
      s1 = __builtin_amdgcn_wmma_f32_16x16x32_bf16(false, aq1, false,
            b_frag_kcontig(lm, ks + kl + 16, kmaxrow, c0 + 32, Dd), (short)0, s1, false, false);
      float p0[8], p1[8];
#pragma unroll
      for (int v = 0; v < 8; ++v) {
        float x0 = s0[v] * scale, x1 = s1[v] * scale;
        if (kl + nn >= eff)      x0 = -1e30f;
        if (kl + 16 + nn >= eff) x1 = -1e30f;
        float tm = fmaxf(x0, x1);
        tm = fmaxf(tm, __shfl_xor(tm, 1)); tm = fmaxf(tm, __shfl_xor(tm, 2));
        tm = fmaxf(tm, __shfl_xor(tm, 4)); tm = fmaxf(tm, __shfl_xor(tm, 8));
        const float nm = fmaxf(mr[v], tm);
        const float al = __expf(mr[v] - nm);
        const float e0 = __expf(x0 - nm), e1 = __expf(x1 - nm);
        float rs = e0 + e1;
        rs += __shfl_xor(rs, 1); rs += __shfl_xor(rs, 2);
        rs += __shfl_xor(rs, 4); rs += __shfl_xor(rs, 8);
        lr[v] = lr[v] * al + rs; mr[v] = nm;
        p0[v] = e0; p1[v] = e1;
#pragma unroll
        for (int t = 0; t < 4; ++t) o[t][v] *= al;
      }
#pragma unroll
      for (int v = 0; v < 8; ++v) {
        const int r = v + half * 8;
        myP[r * 32 + nn]      = f2bf(p0[v]);
        myP[r * 32 + 16 + nn] = f2bf(p1[v]);
      }
      bf16x16 bv[4];
#pragma unroll
      for (int t = 0; t < 4; ++t) bv[t] = b_frag_kstrided(lm, ks + kl, kmaxrow, c0 + t * 16, Dd);
      bf16x16 pa;
      {
        const int m = lane & 15, kb2 = (lane >> 4) << 4;
#pragma unroll
        for (int e = 0; e < 8; ++e) pa[e] = myP[m * 32 + kb2 + e];
#pragma unroll
        for (int e = 0; e < 8; ++e) pa[8 + e] = myP[m * 32 + kb2 + 8 + e];
      }
#pragma unroll
      for (int t = 0; t < 4; ++t)
        o[t] = __builtin_amdgcn_wmma_f32_16x16x32_bf16(false, pa, false, bv[t],
                                                       (short)0, o[t], false, false);
    }
#pragma unroll
    for (int v = 0; v < 8; ++v) {
      const int qrow = q0 + v + half * 8;
      if (qrow - st < eff) {
        const float inv = 1.f / lr[v];
#pragma unroll
        for (int t = 0; t < 4; ++t)
          cmb[(size_t)qrow * (2 * Dd) + c0 + t * 16 + nn] = f2bf(o[t][v] * inv);
      }
    }
  }
}

// ---------------------------------------------------------------------------
extern "C" void kernel_launch(void* const* d_in, const int* in_sizes, int n_in,
                              void* d_out, int out_size, void* d_ws, size_t ws_size,
                              hipStream_t stream) {
  (void)in_sizes; (void)n_in; (void)out_size; (void)ws_size;
  const float* x    = (const float*)d_in[0];
  const float* ga   = (const float*)d_in[1];
  const float* ba   = (const float*)d_in[2];
  const float* gb   = (const float*)d_in[3];
  const float* bb   = (const float*)d_in[4];
  const float* Wq   = (const float*)d_in[5];
  const float* bq   = (const float*)d_in[6];
  const float* Wk   = (const float*)d_in[7];
  const float* Wv   = (const float*)d_in[8];
  const float* bv   = (const float*)d_in[9];
  const float* Wo   = (const float*)d_in[10];
  const float* bo   = (const float*)d_in[11];
  const float* Wp   = (const float*)d_in[12];
  const float* bp   = (const float*)d_in[13];
  const float* Wpj  = (const float*)d_in[14];
  const float* bpj  = (const float*)d_in[15];
  float* out = (float*)d_out;

  char* ws = (char*)d_ws;
  size_t off = 0;
  auto alloc = [&](size_t bytes) -> void* {
    void* p = ws + off;
    off += (bytes + 255) & ~(size_t)255;
    return p;
  };
  const size_t MD = (size_t)M_ROWS * Dd;
  __bf16* localbf = (__bf16*)alloc(MD * 2);
  __bf16* globebf = (__bf16*)alloc(MD * 2);
  __bf16* Wqb  = (__bf16*)alloc((size_t)Dd * Dd * 2);
  __bf16* Wkb  = (__bf16*)alloc((size_t)Dd * Dd * 2);
  __bf16* Wvb  = (__bf16*)alloc((size_t)Dd * Dd * 2);
  __bf16* Wob  = (__bf16*)alloc((size_t)Dd * Dd * 2);
  __bf16* Wpjb = (__bf16*)alloc((size_t)Dd * 2 * Dd * 2);
  __bf16* qbf  = (__bf16*)alloc(MD * 2);
  __bf16* kbf  = (__bf16*)alloc(MD * 2);
  __bf16* vbf  = (__bf16*)alloc(MD * 2);
  __bf16* abf  = (__bf16*)alloc(MD * 2);
  float*  gout = (float*)alloc(MD * 4);
  __bf16* comb = (__bf16*)alloc(MD * 2 * 2);          // [B,S,2D] bf16
  float*  cmean = (float*)alloc((size_t)Bb * Dd * 4);
  float*  pf = (float*)alloc(64);
  int*    pi = (int*)alloc(64);

  ln_kernel<<<M_ROWS, 256, 0, stream>>>(x, ga, ba, gb, bb, localbf, globebf);

  cvt_bf16_kernel<<<(Dd * Dd) / 256, 256, 0, stream>>>(Wq, Wqb, Dd * Dd);
  cvt_bf16_kernel<<<(Dd * Dd) / 256, 256, 0, stream>>>(Wk, Wkb, Dd * Dd);
  cvt_bf16_kernel<<<(Dd * Dd) / 256, 256, 0, stream>>>(Wv, Wvb, Dd * Dd);
  cvt_bf16_kernel<<<(Dd * Dd) / 256, 256, 0, stream>>>(Wo, Wob, Dd * Dd);
  cvt_bf16_kernel<<<(Dd * 2 * Dd) / 256, 256, 0, stream>>>(Wpj, Wpjb, Dd * 2 * Dd);

  dim3 ggrid(Dd / 64, M_ROWS / 128);
  gemm_bf16_kernel<Dd><<<ggrid, 256, 0, stream>>>(globebf, Dd, Wqb, Dd, bq,
                                                  nullptr, 0, qbf, Dd);
  gemm_bf16_kernel<Dd><<<ggrid, 256, 0, stream>>>(globebf, Dd, Wkb, Dd, nullptr,
                                                  nullptr, 0, kbf, Dd);
  gemm_bf16_kernel<Dd><<<ggrid, 256, 0, stream>>>(globebf, Dd, Wvb, Dd, bv,
                                                  nullptr, 0, vbf, Dd);

  attn_global_kernel<<<dim3(Ss / 128, Bb * Hh), 256, 0, stream>>>(qbf, kbf, vbf, abf);

  gemm_bf16_kernel<Dd><<<ggrid, 256, 0, stream>>>(abf, Dd, Wob, Dd, bo,
                                                  gout, Dd, comb + Dd, 2 * Dd);

  colmean_kernel<<<(Bb * Dd) / 256, 256, 0, stream>>>(gout, cmean);
  predict_kernel<<<1, 256, 0, stream>>>(cmean, Wp, bp, pf, pi);

  attn_local_kernel<<<dim3(Ss, Bb * Hh), 256, 0, stream>>>(localbf, comb, pf, pi);

  gemm_bf16_kernel<2 * Dd><<<ggrid, 256, 0, stream>>>(comb, 2 * Dd, Wpjb, 2 * Dd, bpj,
                                                      out, Dd, nullptr, 0);
}
